// EnhancedMultiHeadAttention_67662914781803
// MI455X (gfx1250) — matrix operand order repaired and checked by reading, not compiled, not run on hardware
//
#include <hip/hip_runtime.h>
#include <hip/hip_bf16.h>
#include <cstdint>
#include <cstddef>

typedef __attribute__((ext_vector_type(16))) __bf16 v16bf;
typedef __attribute__((ext_vector_type(8)))  float  v8f;

#define DEV static __device__ __forceinline__

// ---------- helpers ----------

DEV __bf16 f2bf(float f) {
  // round-to-nearest-even fp32 -> bf16 via bit twiddle
  uint32_t u = __builtin_bit_cast(uint32_t, f);
  uint32_t r = u + 0x7FFFu + ((u >> 16) & 1u);
  unsigned short h = (unsigned short)(r >> 16);
  return __builtin_bit_cast(__bf16, h);
}

DEV v8f vzero8() { v8f z = {0.f,0.f,0.f,0.f,0.f,0.f,0.f,0.f}; return z; }

DEV v8f wmma_bf16(v16bf a, v16bf b, v8f c) {
  // D = A(16x32 bf16) * B(32x16 bf16) + C(16x16 f32)
  return __builtin_amdgcn_wmma_f32_16x16x32_bf16(false, a, false, b, (short)0, c, false, false);
}

// A fragment (16x32, bf16): lane<16 holds row=lane, K = {0..7,16..23};
// lane>=16 holds row=lane-16, K = {8..15,24..31}. Pairs packed per 32-bit reg.
DEV v16bf load_afrag(const __bf16* p, int ld) {
  int lane = threadIdx.x & 31;
  const __bf16* row = p + (size_t)(lane & 15) * ld;
  int off = (lane < 16) ? 0 : 8;
  union { v16bf v; uint32_t u[8]; } r;
#pragma unroll
  for (int i = 0; i < 4; ++i) {
    r.u[i]     = *(const uint32_t*)(row + off + 2 * i);
    r.u[4 + i] = *(const uint32_t*)(row + off + 16 + 2 * i);
  }
  return r.v;
}

// B fragment (32x16, bf16): lane holds column n=lane&15 (a row of the K-major
// source matrix); lane<16 covers K={0..7,16..23}, lane>=16 covers K={8..15,24..31}.
DEV v16bf load_bfrag(const __bf16* p, int ld) {
  int lane = threadIdx.x & 31;
  const __bf16* row = p + (size_t)(lane & 15) * ld;
  int off = (lane < 16) ? 0 : 8;
  union { v16bf v; uint32_t u[8]; } r;
#pragma unroll
  for (int i = 0; i < 4; ++i) {
    r.u[i]     = *(const uint32_t*)(row + off + 2 * i);
    r.u[4 + i] = *(const uint32_t*)(row + off + 16 + 2 * i);
  }
  return r.v;
}

// Per-lane async 16B copy global -> LDS (CDNA5, tracked by ASYNCcnt).
DEV void async_copy_b128(uint32_t lds_byte_addr, uint32_t gbyte_off, uint64_t gbase) {
  asm volatile("global_load_async_to_lds_b128 %0, %1, %2"
               :: "v"(lds_byte_addr), "v"(gbyte_off), "s"(gbase) : "memory");
}

DEV void wait_async0() {
  asm volatile("s_wait_asynccnt 0" ::: "memory");
}

// ---------- small kernels ----------

__global__ void mask_table_kernel(float* tab) {
  int d = blockIdx.x * blockDim.x + threadIdx.x;
  if (d < 2048) {
    float fd = (float)d;
    float d2 = fd * fd;
    tab[d] = 0.25f * (__expf(-d2 * (1.0f / 50.0f))  + __expf(-d2 * (1.0f / 200.0f)) +
                      __expf(-d2 * (1.0f / 800.0f)) + __expf(-d2 * (1.0f / 3200.0f)));
  }
}

__global__ void cvt_f32_bf16_kernel(const float* __restrict__ x, __bf16* __restrict__ y, int n) {
  int i = blockIdx.x * blockDim.x + threadIdx.x;
  int stride = gridDim.x * blockDim.x;
  for (; i < n; i += stride) y[i] = f2bf(x[i]);
}

// ---------- GEMM:  Y[m,n] = (sum_k A[m,k] * W[n,k] + bias[n]) * scale ----------
// M=4096, N=1024, K=1024.  Block tile 64(M) x 128(N), 8 waves of 32x32.
// A/B tiles double-buffered in LDS via async global->LDS b128 copies.
// mode 0: bf16 out to [B,H,S,dk]; mode 1: bf16 out to [B,H,dk,S]; mode 2: fp32 [M,N].
#define LDPAD 40   // padded LDS row stride in elements (80B: 16B-aligned, conflict-free)

__global__ void __launch_bounds__(256)
gemm_wmma_kernel(const __bf16* __restrict__ A, const __bf16* __restrict__ W,
                 const float* __restrict__ bias,
                 __bf16* __restrict__ outB, float* __restrict__ outF,
                 int mode, float scale) {
  __shared__ __bf16 ldsA[2][64 * LDPAD];
  __shared__ __bf16 ldsB[2][128 * LDPAD];

  int tid  = threadIdx.x;
  int wid  = tid >> 5;
  int lane = tid & 31;
  int bm = blockIdx.x >> 3;    // 64 M-block tiles
  int bn = blockIdx.x & 7;     // 8  N-block tiles
  int wm = wid >> 2;           // 0..1
  int wn = wid & 3;            // 0..3
  int m0 = bm * 64 + wm * 32;
  int n0 = bn * 128 + wn * 32;
  int hi8 = (lane < 16) ? 0 : 8;
  int nl = lane & 15;

  uint64_t gA = (uint64_t)(size_t)A;
  uint64_t gW = (uint64_t)(size_t)W;

  // A-tile: 64 rows x 32 k (4 x 16B chunks/row) = 256 chunks -> 1 per thread
  int arow = tid >> 2, acp = tid & 3;
  uint32_t aLdsBase = (uint32_t)(size_t)(&ldsA[0][0]);
  uint32_t bLdsBase = (uint32_t)(size_t)(&ldsB[0][0]);
  uint32_t aLdsOff = (uint32_t)(arow * LDPAD * 2 + acp * 16);
  uint32_t aGOff0  = (uint32_t)(((bm * 64 + arow) * 1024) * 2 + acp * 16);
  // B-tile: 128 rows x 32 k = 512 chunks -> 2 per thread
  int brow0 = tid >> 2,          bcp0 = tid & 3;
  int brow1 = (tid + 256) >> 2,  bcp1 = tid & 3;
  uint32_t bLdsOff0 = (uint32_t)(brow0 * LDPAD * 2 + bcp0 * 16);
  uint32_t bLdsOff1 = (uint32_t)(brow1 * LDPAD * 2 + bcp1 * 16);
  uint32_t bGOff0   = (uint32_t)(((bn * 128 + brow0) * 1024) * 2 + bcp0 * 16);
  uint32_t bGOff1   = (uint32_t)(((bn * 128 + brow1) * 1024) * 2 + bcp1 * 16);
  const uint32_t bufStrideA = (uint32_t)(64 * LDPAD * 2);
  const uint32_t bufStrideB = (uint32_t)(128 * LDPAD * 2);

  v8f acc00 = vzero8(), acc01 = vzero8(), acc10 = vzero8(), acc11 = vzero8();

  // prologue: stage k-step 0 into buffer 0
  async_copy_b128(aLdsBase + aLdsOff, aGOff0, gA);
  async_copy_b128(bLdsBase + bLdsOff0, bGOff0, gW);
  async_copy_b128(bLdsBase + bLdsOff1, bGOff1, gW);

  for (int it = 0; it < 32; ++it) {
    int cur = it & 1;
    wait_async0();
    __syncthreads();
    if (it + 1 < 32) {
      int nxt = cur ^ 1;
      uint32_t kb = (uint32_t)((it + 1) * 32 * 2);   // k byte offset
      async_copy_b128(aLdsBase + nxt * bufStrideA + aLdsOff, aGOff0 + kb, gA);
      async_copy_b128(bLdsBase + nxt * bufStrideB + bLdsOff0, bGOff0 + kb, gW);
      async_copy_b128(bLdsBase + nxt * bufStrideB + bLdsOff1, bGOff1 + kb, gW);
    }
    const __bf16* la = &ldsA[cur][0];
    const __bf16* lb = &ldsB[cur][0];
    v16bf a0 = load_afrag(la + (wm * 32) * LDPAD, LDPAD);
    v16bf a1 = load_afrag(la + (wm * 32 + 16) * LDPAD, LDPAD);
    v16bf b0 = load_bfrag(lb + (wn * 32) * LDPAD, LDPAD);
    v16bf b1 = load_bfrag(lb + (wn * 32 + 16) * LDPAD, LDPAD);
    acc00 = wmma_bf16(a0, b0, acc00);
    acc01 = wmma_bf16(a0, b1, acc01);
    acc10 = wmma_bf16(a1, b0, acc10);
    acc11 = wmma_bf16(a1, b1, acc11);
  }

#pragma unroll
  for (int ti = 0; ti < 2; ++ti) {
#pragma unroll
    for (int tj = 0; tj < 2; ++tj) {
      v8f a = (ti == 0) ? (tj == 0 ? acc00 : acc01) : (tj == 0 ? acc10 : acc11);
      int n = n0 + tj * 16 + nl;
      float bb = bias[n];
#pragma unroll
      for (int r = 0; r < 8; ++r) {
        int m = m0 + ti * 16 + hi8 + r;
        float val = (a[r] + bb) * scale;
        if (mode == 2) {
          outF[(size_t)m * 1024 + n] = val;
        } else {
          int bi = m >> 11, s = m & 2047, hh = n >> 6, d = n & 63;
          size_t idx;
          if (mode == 0) idx = (((size_t)(bi * 16 + hh) * 2048 + s) * 64 + d);
          else           idx = (((size_t)(bi * 16 + hh) * 64 + d) * 2048 + s);
          outB[idx] = f2bf(val);
        }
      }
    }
  }
}

// ---------- flash attention ----------
// qh/kh: bf16 [B,H,S,dk] (q pre-scaled by 1/sqrt(dk)); vt: bf16 [B,H,dk,S].
// ctx out: bf16 [B*S, H*dk] row-major (input to final projection).
__global__ void __launch_bounds__(256)
attn_wmma_kernel(const __bf16* __restrict__ qh, const __bf16* __restrict__ kh,
                 const __bf16* __restrict__ vt, const float* __restrict__ maskTab,
                 __bf16* __restrict__ ctx) {
  __shared__ __bf16 Pst[8 * 16 * 36];     // per-wave 16x(32 + 4 pad) P staging
  int wid  = threadIdx.x >> 5;
  int lane = threadIdx.x & 31;
  int wg = blockIdx.x * 8 + wid;          // 4096 wave tiles
  int bh = wg >> 7;                        // 32 (b,h) pairs
  int qt = wg & 127;                       // 128 q tiles of 16 rows
  int q0 = qt * 16;
  int b = bh >> 4, h = bh & 15;
  const __bf16* qp = qh + (size_t)bh * 2048 * 64;
  const __bf16* kp = kh + (size_t)bh * 2048 * 64;
  const __bf16* vp = vt + (size_t)bh * 64 * 2048;
  __bf16* pl = Pst + wid * (16 * 36);
  int hi8 = (lane < 16) ? 0 : 8;
  int nl = lane & 15;

  // q A-fragments for dk 0..31 and 32..63, reused for all key tiles
  v16bf aqlo = load_afrag(qp + (size_t)q0 * 64, 64);
  v16bf aqhi = load_afrag(qp + (size_t)q0 * 64 + 32, 64);

  v8f o0 = vzero8(), o1 = vzero8(), o2 = vzero8(), o3 = vzero8();
  float mi[8], li[8];
#pragma unroll
  for (int r = 0; r < 8; ++r) { mi[r] = -3.0e38f; li[r] = 0.f; }

  for (int k0 = 0; k0 < 2048; k0 += 32) {
    // scores for 32 keys: two 16x16 tiles, K-dim = dk = 64 (2 WMMA steps each)
    v16bf b0 = load_bfrag(kp + (size_t)k0 * 64, 64);
    v16bf b1 = load_bfrag(kp + (size_t)(k0 + 16) * 64, 64);
    v8f s0 = wmma_bf16(aqlo, b0, vzero8());
    v8f s1 = wmma_bf16(aqlo, b1, vzero8());
    b0 = load_bfrag(kp + (size_t)k0 * 64 + 32, 64);
    b1 = load_bfrag(kp + (size_t)(k0 + 16) * 64 + 32, 64);
    s0 = wmma_bf16(aqhi, b0, s0);
    s1 = wmma_bf16(aqhi, b1, s1);

    float alpha[8];
#pragma unroll
    for (int r = 0; r < 8; ++r) {
      int m = hi8 + r;
      int i = q0 + m;
      int j0 = k0 + nl;
      float sv0 = s0[r];
      float sv1 = s1[r];
      int d0 = i - j0;        if (d0 >= 0) sv0 += maskTab[d0];
      int d1 = i - j0 - 16;   if (d1 >= 0) sv1 += maskTab[d1];
      // row max over this tile (16 lanes per half-wave share a row)
      float t = fmaxf(sv0, sv1);
      t = fmaxf(t, __shfl_xor(t, 1));
      t = fmaxf(t, __shfl_xor(t, 2));
      t = fmaxf(t, __shfl_xor(t, 4));
      t = fmaxf(t, __shfl_xor(t, 8));
      float mn = fmaxf(mi[r], t);
      float al = __expf(mi[r] - mn);
      mi[r] = mn;
      float p0 = __expf(sv0 - mn);
      float p1 = __expf(sv1 - mn);
      float rs = p0 + p1;
      rs += __shfl_xor(rs, 1);
      rs += __shfl_xor(rs, 2);
      rs += __shfl_xor(rs, 4);
      rs += __shfl_xor(rs, 8);
      li[r] = li[r] * al + rs;
      alpha[r] = al;
      // stage P (bf16) to LDS in plain row-major for D->A relayout
      pl[m * 36 + nl]      = f2bf(p0);
      pl[m * 36 + nl + 16] = f2bf(p1);
    }
#pragma unroll
    for (int r = 0; r < 8; ++r) {
      o0[r] *= alpha[r]; o1[r] *= alpha[r]; o2[r] *= alpha[r]; o3[r] *= alpha[r];
    }

    // P as A operand (16x32), V^T columns as B operand (contiguous key pairs)
    v16bf ap = load_afrag(pl, 36);
    v16bf bv = load_bfrag(vp + (size_t)0 * 16 * 2048 + k0, 2048);
    o0 = wmma_bf16(ap, bv, o0);
    bv = load_bfrag(vp + (size_t)1 * 16 * 2048 + k0, 2048);
    o1 = wmma_bf16(ap, bv, o1);
    bv = load_bfrag(vp + (size_t)2 * 16 * 2048 + k0, 2048);
    o2 = wmma_bf16(ap, bv, o2);
    bv = load_bfrag(vp + (size_t)3 * 16 * 2048 + k0, 2048);
    o3 = wmma_bf16(ap, bv, o3);
  }

#pragma unroll
  for (int r = 0; r < 8; ++r) {
    float inv = 1.0f / li[r];
    o0[r] *= inv; o1[r] *= inv; o2[r] *= inv; o3[r] *= inv;
  }
#pragma unroll
  for (int r = 0; r < 8; ++r) {
    int s = q0 + hi8 + r;
    size_t rowbase = ((size_t)(b * 2048 + s)) * 1024 + h * 64 + nl;
    ctx[rowbase + 0]  = f2bf(o0[r]);
    ctx[rowbase + 16] = f2bf(o1[r]);
    ctx[rowbase + 32] = f2bf(o2[r]);
    ctx[rowbase + 48] = f2bf(o3[r]);
  }
}

// ---------- launch ----------

extern "C" void kernel_launch(void* const* d_in, const int* in_sizes, int n_in,
                              void* d_out, int out_size, void* d_ws, size_t ws_size,
                              hipStream_t stream) {
  const float* Q  = (const float*)d_in[0];
  const float* K  = (const float*)d_in[1];
  const float* V  = (const float*)d_in[2];
  const float* Wq = (const float*)d_in[3];
  const float* bq = (const float*)d_in[4];
  const float* Wk = (const float*)d_in[5];
  const float* bk = (const float*)d_in[6];
  const float* Wv = (const float*)d_in[7];
  const float* bv = (const float*)d_in[8];
  const float* Wo = (const float*)d_in[9];
  const float* bo = (const float*)d_in[10];
  float* out = (float*)d_out;

  const size_t MK = (size_t)4096 * 1024;  // activation elems
  const size_t KK = (size_t)1024 * 1024;  // weight elems

  char* w = (char*)d_ws;
  float*  maskTab = (float*)w;  w += 4096 * sizeof(float);
  __bf16* Qb  = (__bf16*)w;     w += MK * 2;
  __bf16* Kb  = (__bf16*)w;     w += MK * 2;
  __bf16* Vb  = (__bf16*)w;     w += MK * 2;
  __bf16* Wqb = (__bf16*)w;     w += KK * 2;
  __bf16* Wkb = (__bf16*)w;     w += KK * 2;
  __bf16* Wvb = (__bf16*)w;     w += KK * 2;
  __bf16* Wob = (__bf16*)w;     w += KK * 2;
  __bf16* qhd = (__bf16*)w;     w += MK * 2;   // [B,H,S,dk], pre-scaled
  __bf16* khd = (__bf16*)w;     w += MK * 2;   // [B,H,S,dk]
  __bf16* vtd = (__bf16*)w;     w += MK * 2;   // [B,H,dk,S]
  __bf16* ctx = (__bf16*)w;     w += MK * 2;   // [B*S, H*dk]

  mask_table_kernel<<<8, 256, 0, stream>>>(maskTab);
  cvt_f32_bf16_kernel<<<2048, 256, 0, stream>>>(Q, Qb, (int)MK);
  cvt_f32_bf16_kernel<<<2048, 256, 0, stream>>>(K, Kb, (int)MK);
  cvt_f32_bf16_kernel<<<2048, 256, 0, stream>>>(V, Vb, (int)MK);
  cvt_f32_bf16_kernel<<<1024, 256, 0, stream>>>(Wq, Wqb, (int)KK);
  cvt_f32_bf16_kernel<<<1024, 256, 0, stream>>>(Wk, Wkb, (int)KK);
  cvt_f32_bf16_kernel<<<1024, 256, 0, stream>>>(Wv, Wvb, (int)KK);
  cvt_f32_bf16_kernel<<<1024, 256, 0, stream>>>(Wo, Wob, (int)KK);

  // projections; q folds 1/sqrt(64)
  gemm_wmma_kernel<<<512, 256, 0, stream>>>(Qb, Wqb, bq, qhd, nullptr, 0, 0.125f);
  gemm_wmma_kernel<<<512, 256, 0, stream>>>(Kb, Wkb, bk, khd, nullptr, 0, 1.0f);
  gemm_wmma_kernel<<<512, 256, 0, stream>>>(Vb, Wvb, bv, vtd, nullptr, 1, 1.0f);

  attn_wmma_kernel<<<512, 256, 0, stream>>>(qhd, khd, vtd, maskTab, ctx);

  // output projection: ctx (bf16) @ Wo^T + bo -> fp32 d_out
  gemm_wmma_kernel<<<512, 256, 0, stream>>>(ctx, Wob, bo, nullptr, out, 2, 1.0f);
}
